// Self_Attn_17068200034621
// MI455X (gfx1250) — matrix operand order repaired and checked by reading, not compiled, not run on hardware
//
#include <hip/hip_runtime.h>

// ---------------------------------------------------------------------------
// Self-attention (SAGAN-style), MI455X / gfx1250, wave32, WMMA bf16.
//   B=4, C=256, CR=32, H=W=64, N=4096.
//   Phase 1: bf16 projections  Q,K:[B,N,32]  V:[B,C,N]   (workspace)
//   Phase 2: flash-attention, transposed dataflow:
//            S^T = K·Q^T  (softmax column = one lane -> cheap reductions)
//            O^T = V^T·P^T (accumulated in f32 C-fragments)
//   K tile streams via GLOBAL_LOAD_ASYNC_TO_LDS_B128 (ASYNCcnt);
//   V tile streams via TENSOR_LOAD_TO_LDS (TDM, TENSORcnt, HW LDS padding).
// ---------------------------------------------------------------------------

typedef __bf16 bf16;
typedef __attribute__((ext_vector_type(16))) __bf16 v16bf;
typedef __attribute__((ext_vector_type(8)))  float  v8f;
typedef __attribute__((ext_vector_type(4)))  unsigned int u32x4;
typedef __attribute__((ext_vector_type(8)))  int i32x8;
typedef __attribute__((ext_vector_type(4)))  int i32x4;

#define BATCH 4
#define CDIM  256
#define CRDIM 32
#define NPOS  4096

union BfFrag  { uint4 u[2]; v16bf v; };
union BfPack8 { bf16 h[8]; uint4 u; };

// Load a 16-element bf16 fragment from two 16-byte chunks (A / B layouts).
__device__ inline v16bf load_frag2(const bf16* p0, const bf16* p1) {
    BfFrag f;
    f.u[0] = *(const uint4*)p0;
    f.u[1] = *(const uint4*)p1;
    return f.v;
}

// Async memory->LDS copy of 16 bytes (per lane), tracked by ASYNCcnt.
// LDS byte offset = low 32 bits of the generic pointer (LDS aperture layout).
__device__ inline void async_copy_b128(void* lds_dst, const void* gsrc) {
    unsigned int       ldsoff = (unsigned int)(unsigned long long)lds_dst;
    unsigned long long gaddr  = (unsigned long long)gsrc;
    asm volatile("global_load_async_to_lds_b128 %0, %1, off"
                 :: "v"(ldsoff), "v"(gaddr) : "memory");
}
__device__ inline void async_wait0() {
    asm volatile("s_wait_asynccnt 0x0" ::: "memory");
}

// TDM 2D tile load, memory -> LDS, bf16 elements, with hardware LDS padding:
// rows of 64 elems (128B = 32 DWORDs, pad_interval code 4) followed by 16B
// (4 DWORDs, pad_amount code 3) of pad -> LDS row stride 72 halfs.
// D# per CDNA5 ISA 8.3/8.4; groups 2/3 zero (2D tensor).
// This toolchain's builtin is the 6-arg flavor:
//   (u32x4 g0, i32x8 g1, i32x4 g2, i32x4 g3, i32x8 g4, i32 cpol)
__device__ inline void tdm_load_v_tile(void* lds_dst, const void* gsrc,
                                       unsigned tile_rows) {
    unsigned long long ga = (unsigned long long)gsrc;
    u32x4 g0;
    g0.x = 1u;                                       // count=1 (valid user D#)
    g0.y = (unsigned int)(unsigned long long)lds_dst; // lds_addr (bytes)
    g0.z = (unsigned int)(ga & 0xffffffffu);          // global_addr[31:0]
    g0.w = (unsigned int)((ga >> 32) & 0x01ffffffu)   // global_addr[56:32]
         | (2u << 30);                                // type = 2 ("image")
    i32x8 g1;
    g1[0] = (int)((1u << 16)        // data_size = 1 -> 2 bytes
                | (1u << 20)        // pad_enable
                | (4u << 22)        // pad_interval: 32 DWORDs per row
                | (3u << 25));      // pad_amount: 4 DWORDs
    g1[1] = (int)((NPOS & 0xffffu) << 16);                   // tensor_dim0 lo16
    g1[2] = (int)((NPOS >> 16) | ((CDIM & 0xffffu) << 16));  // dim0 hi | dim1 lo
    g1[3] = (int)((64u & 0xffffu) << 16);                    // dim1 hi=0 | tile_dim0=64
    g1[4] = (int)(tile_rows & 0xffffu);                      // tile_dim1 (tile_dim2=0)
    g1[5] = (int)NPOS;                                       // tensor_dim0_stride lo32
    g1[6] = 0;                                               // stride hi | dim1_stride lo
    g1[7] = 0;
    const i32x4 zz4 = (i32x4){0, 0, 0, 0};
    const i32x8 zz8 = (i32x8){0, 0, 0, 0, 0, 0, 0, 0};
    __builtin_amdgcn_tensor_load_to_lds(g0, g1, zz4, zz4, zz8, 0);
}

// ---------------------------------------------------------------------------
// Kernel 1: projections.  One block = one batch b, one 32-position tile.
//   xs tile [256 c][32 n] f32 in LDS (stride 36 floats keeps b128 alignment).
// ---------------------------------------------------------------------------
__global__ __launch_bounds__(256) void proj_kernel(
    const float* __restrict__ x,
    const float* __restrict__ Wq, const float* __restrict__ bq,
    const float* __restrict__ Wk, const float* __restrict__ bk,
    const float* __restrict__ Wv, const float* __restrict__ bv,
    bf16* __restrict__ Qw, bf16* __restrict__ Kw, bf16* __restrict__ Vw)
{
    __shared__ float xs[256 * 36];

    const int t  = threadIdx.x;
    const int b  = blockIdx.x >> 7;          // / 128 tiles
    const int n0 = (blockIdx.x & 127) * 32;

    // ---- load x tile: thread t loads channel row c = t (contiguous 128B) ----
    {
        const float* src = x + ((size_t)(b * CDIM + t) * NPOS) + n0;
        float* dst = xs + t * 36;
        #pragma unroll
        for (int j = 0; j < 32; j += 4)
            *(float4*)(dst + j) = *(const float4*)(src + j);
    }
    __syncthreads();

    // ---- V projection: thread t = output channel, 32 positions ----
    {
        float acc[32];
        #pragma unroll
        for (int j = 0; j < 32; ++j) acc[j] = 0.f;
        const float* wr = Wv + t * CDIM;
        for (int c = 0; c < CDIM; ++c) {
            const float w = wr[c];
            const float4* xr4 = (const float4*)(xs + c * 36);
            #pragma unroll
            for (int jq = 0; jq < 8; ++jq) {
                float4 v = xr4[jq];                 // LDS broadcast read
                acc[4*jq+0] += w * v.x;
                acc[4*jq+1] += w * v.y;
                acc[4*jq+2] += w * v.z;
                acc[4*jq+3] += w * v.w;
            }
        }
        const float bias = bv[t];
        bf16* dst = Vw + ((size_t)(b * CDIM + t) * NPOS) + n0;
        #pragma unroll
        for (int j = 0; j < 32; ++j) dst[j] = (bf16)(acc[j] + bias);
    }

    // ---- Q/K projections: threads 0..63 (d = t&31, t>=32 -> K) ----
    if (t < 64) {
        const int  d   = t & 31;
        const bool isK = (t >= 32);
        const float* wr   = (isK ? Wk : Wq) + d * CDIM;
        const float  bias = (isK ? bk : bq)[d];
        float acc[32];
        #pragma unroll
        for (int j = 0; j < 32; ++j) acc[j] = 0.f;
        for (int c = 0; c < CDIM; ++c) {
            const float w = wr[c];
            const float4* xr4 = (const float4*)(xs + c * 36);
            #pragma unroll
            for (int jq = 0; jq < 8; ++jq) {
                float4 v = xr4[jq];
                acc[4*jq+0] += w * v.x;
                acc[4*jq+1] += w * v.y;
                acc[4*jq+2] += w * v.z;
                acc[4*jq+3] += w * v.w;
            }
        }
        bf16* base = (isK ? Kw : Qw) + ((size_t)b * NPOS + n0) * CRDIM + d;
        #pragma unroll
        for (int j = 0; j < 32; ++j) base[(size_t)j * CRDIM] = (bf16)(acc[j] + bias);
    }
}

// ---------------------------------------------------------------------------
// Kernel 2: flash attention (transposed dataflow, one softmax step / 64 j).
//   Block = 8 waves; wave w owns 16 query rows (128 q-rows / block).
//   grid = B * N/128 = 128 blocks.
//   Per 64-j tile:
//     K tile:  async b128 copies (ASYNCcnt)
//     V tile:  one TDM tensor load per wave, 32-channel slice (TENSORcnt)
//     4 S^T WMMAs -> one online-softmax update -> one O rescale
//     -> 4 packed P^T stores -> 2 P^T B-frags -> 32 O^T WMMAs.
// ---------------------------------------------------------------------------
__global__ __launch_bounds__(256) void attn_kernel(
    const bf16* __restrict__ Qw, const bf16* __restrict__ Kw,
    const bf16* __restrict__ Vw,
    const float* __restrict__ x, const float* __restrict__ gamma,
    float* __restrict__ out)
{
    __shared__ bf16 Kt[64 * 32];          //  4 KB: K rows [j][d]
    __shared__ bf16 Vt[256 * 72];         // 36 KB: V transposed [c][j], pad 72
    __shared__ bf16 Ps[8 * 16 * 64];      // 16 KB: per-wave P^T staging [i][j0..63]

    const int t    = threadIdx.x;
    const int wv   = t >> 5;                       // wave 0..7
    const int lane = t & 31;
    const int b    = blockIdx.x >> 5;              // / 32 q-tiles
    const int q0   = (blockIdx.x & 31) * 128 + wv * 16;

    const int li = lane & 15;                      // query column i / matrix N
    const int hi = lane >> 4;

    // Resident Q^T B-fragment (32d x 16i): lane li = column i = query row,
    // element t -> d = hi*16 + t  (one contiguous 32B chunk of Q row q0+li).
    const bf16* qrow = Qw + ((size_t)(b * NPOS) + q0 + li) * CRDIM + 16 * hi;
    const v16bf qb = load_frag2(qrow, qrow + 8);

    // O^T accumulators: 16 tiles of (16c x 16i); lane holds rows r+8*hi, col li.
    v8f OT[16];
    #pragma unroll
    for (int ct = 0; ct < 16; ++ct) OT[ct] = (v8f){0.f,0.f,0.f,0.f,0.f,0.f,0.f,0.f};
    float mr = -3.0e38f;   // running max for query row q0+li (dup across hi)
    float lr = 0.f;        // running sum

    bf16* psw = Ps + wv * (16 * 64);     // P^T staging: [i][j 0..63], row=128B

    for (int j0 = 0; j0 < NPOS; j0 += 64) {
        __syncthreads();
        // ---- V tile: one TDM load per wave (32-channel slice, HW padding) ----
        tdm_load_v_tile(Vt + (wv * 32) * 72,
                        Vw + ((size_t)(b * CDIM + wv * 32)) * NPOS + j0,
                        /*tile_rows=*/32);
        // ---- K tile: 4 KB, one async b128 per thread (memory -> LDS) ----
        {
            const int row = t >> 2, seg = t & 3;
            const bf16* ksrc = Kw + ((size_t)(b * NPOS) + j0 + row) * CRDIM + seg * 8;
            if (j0 + 64 < NPOS) __builtin_prefetch(ksrc + 64 * CRDIM, 0, 1);
            async_copy_b128(Kt + row * 32 + seg * 8, ksrc);
        }
        async_wait0();                           // s_wait_asynccnt 0
        __builtin_amdgcn_s_wait_tensorcnt(0);    // s_wait_tensorcnt 0
        __syncthreads();

        // ---- 4 S^T sub-tiles (16j x 16i each), K as A-matrix ----
        v8f ST[4];
        const v8f z = (v8f){0.f,0.f,0.f,0.f,0.f,0.f,0.f,0.f};
        #pragma unroll
        for (int s = 0; s < 4; ++s) {
            const bf16* kr = Kt + ((s * 16 + li) * 32);
            const v16bf ka = load_frag2(kr + 16 * hi, kr + 16 * hi + 8);
            ST[s] = __builtin_amdgcn_wmma_f32_16x16x32_bf16(false, ka, false, qb,
                                                            (short)0, z, false, false);
        }

        // ---- one online-softmax step over all 64 j ----
        float mx = fmaxf(ST[0][0], ST[0][1]);
        #pragma unroll
        for (int s = 0; s < 4; ++s)
            #pragma unroll
            for (int r = 0; r < 8; ++r) mx = fmaxf(mx, ST[s][r]);
        mx = fmaxf(mx, __shfl_xor(mx, 16, 32));       // single cross-lane hop
        const float nm = fmaxf(mr, mx);
        const float sc = __expf(mr - nm);
        mr = nm;

        float rs = 0.f;
        #pragma unroll
        for (int s = 0; s < 4; ++s) {
            BfPack8 pk;
            #pragma unroll
            for (int r = 0; r < 8; ++r) {
                const float pe = __expf(ST[s][r] - nm);
                rs += pe;
                pk.h[r] = (bf16)pe;
            }
            // lane holds j = s*16 + 8*hi + r  -> one packed b128 store
            *(uint4*)(psw + li * 64 + s * 16 + 8 * hi) = pk.u;
        }
        rs += __shfl_xor(rs, 16, 32);                 // single cross-lane hop
        lr = lr * sc + rs;

        // one accumulator rescale per 64 j (lane-scalar sc)
        #pragma unroll
        for (int ct = 0; ct < 16; ++ct)
            #pragma unroll
            for (int r = 0; r < 8; ++r) OT[ct][r] *= sc;

        // ---- O^T += V^T (16c x 32j) · P^T (32j x 16i), 2 chunks x 16 WMMAs ----
        #pragma unroll
        for (int jc = 0; jc < 2; ++jc) {
            // P^T B-frag (32j x 16i): contiguous 32B per lane (same-wave LDS
            // ops are in-order, so no barrier needed after the stores above).
            const bf16* prow = psw + li * 64 + jc * 32 + 16 * hi;
            const v16bf pb = load_frag2(prow, prow + 8);
            #pragma unroll
            for (int ct = 0; ct < 16; ++ct) {
                const bf16* vrow = Vt + ((ct * 16 + li) * 72) + jc * 32;
                const v16bf va = load_frag2(vrow + 16 * hi, vrow + 16 * hi + 8);
                OT[ct] = __builtin_amdgcn_wmma_f32_16x16x32_bf16(false, va, false, pb,
                                                                 (short)0, OT[ct], false, false);
            }
        }
    }

    // ---- epilogue: out = gamma * (O^T / l) + x ----
    const float g   = gamma[0];
    const float gil = g / lr;                      // per-lane (query row q0+li)
    #pragma unroll
    for (int ct = 0; ct < 16; ++ct) {
        #pragma unroll
        for (int r = 0; r < 8; ++r) {
            const int c = ct * 16 + r + 8 * hi;    // channel row of O^T
            const size_t idx = ((size_t)(b * CDIM + c)) * NPOS + (q0 + li);
            out[idx] = gil * OT[ct][r] + x[idx];
        }
    }
}

// ---------------------------------------------------------------------------
// Host launcher
// ---------------------------------------------------------------------------
extern "C" void kernel_launch(void* const* d_in, const int* in_sizes, int n_in,
                              void* d_out, int out_size, void* d_ws, size_t ws_size,
                              hipStream_t stream) {
    (void)in_sizes; (void)n_in; (void)out_size; (void)ws_size;
    const float* x     = (const float*)d_in[0];
    const float* Wq    = (const float*)d_in[1];
    const float* bq    = (const float*)d_in[2];
    const float* Wk    = (const float*)d_in[3];
    const float* bk    = (const float*)d_in[4];
    const float* Wv    = (const float*)d_in[5];
    const float* bv    = (const float*)d_in[6];
    const float* gamma = (const float*)d_in[7];
    float* out = (float*)d_out;

    // Workspace: Q [B,N,32] bf16 | K [B,N,32] bf16 | V [B,C,N] bf16  (~10.5 MB)
    bf16* Qw = (bf16*)d_ws;
    bf16* Kw = Qw + (size_t)BATCH * NPOS * CRDIM;
    bf16* Vw = Kw + (size_t)BATCH * NPOS * CRDIM;

    hipLaunchKernelGGL(proj_kernel, dim3(BATCH * (NPOS / 32)), dim3(256), 0, stream,
                       x, Wq, bq, Wk, bk, Wv, bv, Qw, Kw, Vw);
    hipLaunchKernelGGL(attn_kernel, dim3(BATCH * (NPOS / 128)), dim3(256), 0, stream,
                       Qw, Kw, Vw, x, gamma, out);
}
